// EntNet_11888469475689
// MI455X (gfx1250) — matrix operand order repaired and checked by reading, not compiled, not run on hardware
//
#include <hip/hip_runtime.h>
#include <hip/hip_bf16.h>

#define NB      32
#define DD      256
#define VOCAB   100032
#define S_LEN   64
#define Q_LEN   16
#define N_SENT  2048
#define N_Q     128

typedef __attribute__((ext_vector_type(16))) __bf16 v16bf;
typedef __attribute__((ext_vector_type(8)))  float  v8f;

__device__ __forceinline__ float sigm(float x) { return 1.f / (1.f + __expf(-x)); }

// A-matrix (16-bit, 16x32) K offset for (lane, reg): lanes 0-15 hold K 0-7 / 16-23,
// lanes 16-31 hold K 8-15 / 24-31, two packed per VGPR.
__device__ __forceinline__ int a_k0(int lane, int r) {
    return ((lane >> 4) << 3) + ((r < 4) ? (r << 1) : (16 + ((r - 4) << 1)));
}

// ---------------- Kernel 1: story encoder  x[t] = sum_p emb[tok]*f_story[p] ----
__global__ void k_enc_story(const int* __restrict__ inputs, const float* __restrict__ emb,
                            const float* __restrict__ fstory, float* __restrict__ x) {
    int t = blockIdx.x, d = threadIdx.x;
    float acc = 0.f;
    for (int p = 0; p < S_LEN; ++p) {
        int tok = inputs[t * S_LEN + p];
        acc += emb[(size_t)tok * DD + d] * fstory[p * DD + d];
    }
    x[t * DD + d] = acc;
}

// ---------------- Kernel 2: query encoder --------------------------------------
__global__ void k_enc_query(const int* __restrict__ query, const float* __restrict__ emb,
                            const float* __restrict__ fquery, float* __restrict__ qv) {
    int i = blockIdx.x, d = threadIdx.x;
    float acc = 0.f;
    for (int p = 0; p < Q_LEN; ++p) {
        int tok = query[i * Q_LEN + p];
        acc += emb[(size_t)tok * DD + d] * fquery[p * DD + d];
    }
    qv[i * DD + d] = acc;
}

// ---------------- Kernel 3: WS = x @ W^T, KX = x @ keys^T ----------------------
__global__ void k_ws_kx(const float* __restrict__ x, const float* __restrict__ W,
                        const float* __restrict__ keys, float* __restrict__ WS,
                        float* __restrict__ KX) {
    int t = blockIdx.x, d = threadIdx.x;
    __shared__ float sx[DD];
    sx[d] = x[t * DD + d];
    __syncthreads();
    float acc = 0.f;
    const float* wr = W + d * DD;
    for (int k = 0; k < DD; ++k) acc += wr[k] * sx[k];
    WS[t * DD + d] = acc;
    if (d < NB) {
        float a2 = 0.f;
        const float* kr = keys + d * DD;
        for (int k = 0; k < DD; ++k) a2 += kr[k] * sx[k];
        KX[t * NB + d] = a2;
    }
}

// ---------------- Kernel 4: KVb = keys @ V^T + bias ----------------------------
__global__ void k_kvb(const float* __restrict__ keys, const float* __restrict__ V,
                      const float* __restrict__ bias, float* __restrict__ KVb) {
    int j = blockIdx.x, d = threadIdx.x;
    __shared__ float sk[DD];
    sk[d] = keys[j * DD + d];
    __syncthreads();
    float acc = bias[d];
    const float* vr = V + d * DD;
    for (int k = 0; k < DD; ++k) acc += vr[k] * sk[k];
    KVb[j * DD + d] = acc;
}

// ---------------- Kernel 5: sequential memory scan (single workgroup) ----------
// 1024 threads = 32 waves. U lives in LDS as bf16 in WMMA B-layout.
__global__ __launch_bounds__(1024) void k_scan(
        const float* __restrict__ x, const float* __restrict__ WS,
        const float* __restrict__ KX, const float* __restrict__ KVb,
        const float* __restrict__ U, const float* __restrict__ h0,
        const float* __restrict__ a_mem_p,
        float* __restrict__ Hf_ws, float* __restrict__ Hf_out) {
    __shared__ __bf16 Ub[DD * DD];      // 128 KB : 128 chunks (ntile*8+kstep) x 32 lanes x 16 bf16
    __shared__ __bf16 Ha[NB * DD];      //  16 KB : 16 chunks (mtile*8+kstep)
    __shared__ float  Hc[NB * DD];      //  32 KB : current H, row-major f32
    __shared__ float  KVs[NB * DD];     //  32 KB
    __shared__ float  sx[DD];
    __shared__ float  wsr[DD];
    __shared__ float  gs[NB];
    __shared__ float  red[1024];

    const int tid  = threadIdx.x;
    const int wave = tid >> 5, lane = tid & 31;
    const float am = *a_mem_p;

    // ---- setup: H, KVb into LDS -------------------------------------------
    for (int i = tid; i < NB * DD; i += 1024) { Hc[i] = h0[i]; KVs[i] = KVb[i]; }
    // ---- setup: U -> bf16 B-layout (B[k][n] = U[n][k]) --------------------
    for (int it = tid; it < 128 * 32 * 8; it += 1024) {
        int chunk = it >> 8, rem = it & 255, ln = rem >> 3, r = rem & 7;
        int ntile = chunk >> 3, kstep = chunk & 7;
        int n  = ntile * 16 + (ln & 15);
        int k0 = kstep * 32 + a_k0(ln, r);
        int dst = chunk * 512 + ln * 16 + (r << 1);
        Ub[dst]     = (__bf16)U[n * DD + k0];
        Ub[dst + 1] = (__bf16)U[n * DD + k0 + 1];
    }
    __syncthreads();
    // ---- setup: H -> bf16 A-layout ----------------------------------------
    for (int it = tid; it < 16 * 32 * 8; it += 1024) {
        int chunk = it >> 8, rem = it & 255, ln = rem >> 3, r = rem & 7;
        int mtile = chunk >> 3, kstep = chunk & 7;
        int m  = mtile * 16 + (ln & 15);
        int k0 = kstep * 32 + a_k0(ln, r);
        int dst = chunk * 512 + ln * 16 + (r << 1);
        Ha[dst]     = (__bf16)Hc[m * DD + k0];
        Ha[dst + 1] = (__bf16)Hc[m * DD + k0 + 1];
    }

    const int mtile = wave >> 4;    // 0..1
    const int ntile = wave & 15;    // 0..15

    for (int t = 0; t < N_SENT; ++t) {
        __syncthreads();
        // phase A: stage s_t and WS_t
        if (tid < DD)            sx[tid]       = x[t * DD + tid];
        else if (tid < 2 * DD)   wsr[tid - DD] = WS[t * DD + (tid - DD)];
        __syncthreads();
        // phase B: gate g = sigmoid(H@s + KX)
        {
            int j = tid >> 5, seg = tid & 31;
            const float* hr = &Hc[j * DD + seg * 8];
            const float* sr = &sx[seg * 8];
            float a = 0.f;
            #pragma unroll
            for (int e = 0; e < 8; ++e) a += hr[e] * sr[e];
            red[tid] = a;
        }
        __syncthreads();
        if (tid < NB) {
            float a = KX[t * NB + tid];
            for (int s = 0; s < 32; ++s) a += red[tid * 32 + s];
            gs[tid] = sigm(a);
        }
        __syncthreads();
        // phase C: P = H @ U^T via WMMA; epilogue H += g*prelu(P+KVb+Ws)
        {
            v8f acc = {};
            #pragma unroll
            for (int kstep = 0; kstep < 8; ++kstep) {
                const v16bf* Ap = (const v16bf*)&Ha[(mtile * 8 + kstep) * 512 + lane * 16];
                const v16bf* Bp = (const v16bf*)&Ub[(ntile * 8 + kstep) * 512 + lane * 16];
                acc = __builtin_amdgcn_wmma_f32_16x16x32_bf16(
                          false, *Ap, false, *Bp, (short)0, acc, false, false);
            }
            int mbase = mtile * 16 + ((lane >> 4) << 3);
            int n     = ntile * 16 + (lane & 15);
            #pragma unroll
            for (int r = 0; r < 8; ++r) {
                int m = mbase + r;
                float v = acc[r] + KVs[m * DD + n] + wsr[n];
                v = (v >= 0.f) ? v : am * v;
                Hc[m * DD + n] = Hc[m * DD + n] + gs[m] * v;   // owner-exclusive slot
            }
        }
        __syncthreads();
        // phase D: per-row sumsq for L2 norm
        {
            int j = tid >> 5, seg = tid & 31;
            const float* hr = &Hc[j * DD + seg * 8];
            float a = 0.f;
            #pragma unroll
            for (int e = 0; e < 8; ++e) a += hr[e] * hr[e];
            red[tid] = a;
        }
        __syncthreads();
        if (tid < NB) {
            float a = 0.f;
            for (int s = 0; s < 32; ++s) a += red[tid * 32 + s];
            gs[tid] = rsqrtf(a);   // reuse gs as inverse norm
        }
        __syncthreads();
        // phase E: normalize + refresh bf16 A-layout copy
        for (int i = tid; i < NB * DD; i += 1024) Hc[i] *= gs[i >> 8];
        __syncthreads();
        for (int it = tid; it < 16 * 32 * 8; it += 1024) {
            int chunk = it >> 8, rem = it & 255, ln = rem >> 3, r = rem & 7;
            int mt = chunk >> 3, ks = chunk & 7;
            int m  = mt * 16 + (ln & 15);
            int k0 = ks * 32 + a_k0(ln, r);
            int dst = chunk * 512 + ln * 16 + (r << 1);
            Ha[dst]     = (__bf16)Hc[m * DD + k0];
            Ha[dst + 1] = (__bf16)Hc[m * DD + k0 + 1];
        }
    }
    __syncthreads();
    for (int i = tid; i < NB * DD; i += 1024) { Hf_ws[i] = Hc[i]; Hf_out[i] = Hc[i]; }
}

// ---------------- Kernel 6: output module -> p = prelu(q + u@Hw^T + Hb) --------
__global__ void k_outmod(const float* __restrict__ qv, const float* __restrict__ Hf,
                         const float* __restrict__ Hw, const float* __restrict__ Hb,
                         const float* __restrict__ a_out_p, float* __restrict__ p) {
    int i = blockIdx.x, d = threadIdx.x;
    __shared__ float sH[NB * DD];
    __shared__ float sq[DD];
    __shared__ float su[DD];
    __shared__ float rbuf[DD];
    sq[d] = qv[i * DD + d];
    for (int r = d; r < NB * DD; r += DD) sH[r] = Hf[r];
    __syncthreads();
    float u = 0.f;
    for (int j = 0; j < NB; ++j) {
        float z = sq[d] * sH[j * DD + d];
        rbuf[d] = z; __syncthreads();
        for (int off = 128; off > 0; off >>= 1) {
            if (d < off) rbuf[d] = fmaxf(rbuf[d], rbuf[d + off]);
            __syncthreads();
        }
        float mx = rbuf[0]; __syncthreads();
        rbuf[d] = __expf(z - mx); __syncthreads();
        for (int off = 128; off > 0; off >>= 1) {
            if (d < off) rbuf[d] += rbuf[d + off];
            __syncthreads();
        }
        float lse = mx + __logf(rbuf[0]); __syncthreads();
        u += (z - lse) * sH[j * DD + d];
    }
    su[d] = u; __syncthreads();
    float acc = Hb[d];
    const float* hw = Hw + d * DD;
    for (int k = 0; k < DD; ++k) acc += hw[k] * su[k];
    float pv = sq[d] + acc;
    float ao = *a_out_p;
    p[i * DD + d] = (pv >= 0.f) ? pv : ao * pv;
}

// ---------------- Kernel 7: repack p -> bf16 WMMA A-layout ---------------------
__global__ void k_packA(const float* __restrict__ p, unsigned int* __restrict__ Apk) {
    int it = blockIdx.x * 256 + threadIdx.x;      // 0..4095
    int chunk = it >> 8, rem = it & 255, ln = rem >> 3, r = rem & 7;
    int mtile = chunk >> 3, kstep = chunk & 7;
    int m  = mtile * 16 + (ln & 15);
    int k0 = kstep * 32 + a_k0(ln, r);
    unsigned short lo = __builtin_bit_cast(unsigned short, (__bf16)p[m * DD + k0]);
    unsigned short hi = __builtin_bit_cast(unsigned short, (__bf16)p[m * DD + k0 + 1]);
    Apk[chunk * 256 + ln * 8 + r] = (unsigned int)lo | ((unsigned int)hi << 16);
}

// ---------------- Kernel 8: y = p @ Rw^T + Rb  (WMMA, Rw streamed once) --------
__global__ __launch_bounds__(128) void k_final(const unsigned int* __restrict__ Apk,
                                               const float* __restrict__ Rw,
                                               const float* __restrict__ Rb,
                                               float* __restrict__ y) {
    int wave = threadIdx.x >> 5, lane = threadIdx.x & 31;
    int ntile = blockIdx.x * 4 + wave;
    int n0 = ntile * 16;
    int n  = n0 + (lane & 15);
    int khalf = lane >> 4;
    v8f acc[8] = {};
    #pragma unroll
    for (int kstep = 0; kstep < 8; ++kstep) {
        const float* src = Rw + (size_t)n * DD + kstep * 32 + khalf * 8;
        __builtin_prefetch(src + 32, 0, 1);   // global_prefetch_b8 ahead of stream
        v16bf b;
        #pragma unroll
        for (int e = 0; e < 16; ++e) b[e] = (__bf16)src[(e < 8) ? e : (e + 8)];
        #pragma unroll
        for (int mt = 0; mt < 8; ++mt) {
            const v16bf* a = (const v16bf*)&Apk[((mt * 8 + kstep) << 8) + lane * 8];
            acc[mt] = __builtin_amdgcn_wmma_f32_16x16x32_bf16(
                          false, *a, false, b, (short)0, acc[mt], false, false);
        }
    }
    float rb = Rb[n];
    #pragma unroll
    for (int mt = 0; mt < 8; ++mt) {
        int mbase = mt * 16 + (khalf << 3);
        #pragma unroll
        for (int r = 0; r < 8; ++r) {
            y[(size_t)(mbase + r) * VOCAB + n] = acc[mt][r] + rb;
        }
    }
}

extern "C" void kernel_launch(void* const* d_in, const int* in_sizes, int n_in,
                              void* d_out, int out_size, void* d_ws, size_t ws_size,
                              hipStream_t stream) {
    const int*   inputs  = (const int*)d_in[0];
    const int*   query   = (const int*)d_in[1];
    const float* h0      = (const float*)d_in[2];
    const float* emb     = (const float*)d_in[3];
    const float* fstory  = (const float*)d_in[4];
    const float* fquery  = (const float*)d_in[5];
    const float* U       = (const float*)d_in[6];
    const float* V       = (const float*)d_in[7];
    const float* W       = (const float*)d_in[8];
    const float* bias    = (const float*)d_in[9];
    const float* keys    = (const float*)d_in[10];
    const float* a_mem   = (const float*)d_in[11];
    const float* Hw      = (const float*)d_in[12];
    const float* Hb      = (const float*)d_in[13];
    const float* Rw      = (const float*)d_in[14];
    const float* Rb      = (const float*)d_in[15];
    const float* a_out   = (const float*)d_in[16];

    float* ws  = (float*)d_ws;
    float* x   = ws;                       // 2048*256
    float* qv  = x   + (size_t)N_SENT * DD;   // 128*256
    float* WS  = qv  + (size_t)N_Q * DD;      // 2048*256
    float* KX  = WS  + (size_t)N_SENT * DD;   // 2048*32
    float* KVb = KX  + (size_t)N_SENT * NB;   // 32*256
    float* Hf  = KVb + (size_t)NB * DD;       // 32*256
    float* p   = Hf  + (size_t)NB * DD;       // 128*256
    unsigned int* Apk = (unsigned int*)(p + (size_t)N_Q * DD);  // 4096 dwords

    float* y     = (float*)d_out;                       // [128, 100032]
    float* Hfout = y + (size_t)N_Q * VOCAB;             // [8192]

    k_enc_story<<<dim3(N_SENT), dim3(DD), 0, stream>>>(inputs, emb, fstory, x);
    k_enc_query<<<dim3(N_Q),   dim3(DD), 0, stream>>>(query, emb, fquery, qv);
    k_ws_kx   <<<dim3(N_SENT), dim3(DD), 0, stream>>>(x, W, keys, WS, KX);
    k_kvb     <<<dim3(NB),     dim3(DD), 0, stream>>>(keys, V, bias, KVb);
    k_scan    <<<dim3(1),      dim3(1024), 0, stream>>>(x, WS, KX, KVb, U, h0, a_mem,
                                                        Hf, Hfout);
    k_outmod  <<<dim3(N_Q),    dim3(DD), 0, stream>>>(qv, Hf, Hw, Hb, a_out, p);
    k_packA   <<<dim3(16),     dim3(DD), 0, stream>>>(p, Apk);
    k_final   <<<dim3(VOCAB / 64), dim3(128), 0, stream>>>(Apk, Rw, Rb, y);
}
